// MultiHeadAttention_44032004719068
// MI455X (gfx1250) — matrix operand order repaired and checked by reading, not compiled, not run on hardware
//
#include <hip/hip_runtime.h>
#include <hip/hip_bf16.h>
#include <cmath>

// ---------------------------------------------------------------------------
// MI455X (gfx1250) fused MHA: QKV proj (LDS-tiled f16 WMMA) -> flash
// attention (LDS-staged f16 WMMA, online softmax) -> residual + LayerNorm.
// wave32, v_wmma_f32_16x16x32_f16, double-buffered LDS fed by CDNA5
// GLOBAL_LOAD_ASYNC_TO_LDS_B128 (ASYNCcnt), software pipelined.
// ---------------------------------------------------------------------------

typedef _Float16 half_t;
typedef __attribute__((ext_vector_type(16))) _Float16 v16h;
typedef __attribute__((ext_vector_type(4)))  _Float16 v4h;
typedef __attribute__((ext_vector_type(8)))  float    v8f;

#define BB   2
#define SS   2048
#define DIMM 1024
#define HH   16
#define DHH  64
#define NTOK (BB * SS)   // 4096

union HF16 { v16h h; float4 q[2]; };

// A-operand (MxK, 16x32 f16): lane = row (lane%16), half = lane/16.
// Half 0 holds K k0+0..15, half 1 holds K k0+16..31 (linear packing).
__device__ __forceinline__ v16h load_a_frag(const half_t* rowbase, int k0, int hlf) {
    const half_t* p = rowbase + k0 + hlf * 16;
    HF16 u;
    u.q[0] = *(const float4*)(p);
    u.q[1] = *(const float4*)(p + 8);
    return u.h;
}

// B-operand (KxN, 32x16 f16): lane = col (lane%16); half 0 holds K k0+0..15,
// half 1 holds K k0+16..31 (linear packing).
__device__ __forceinline__ v16h load_b_frag(const half_t* colbase, int k0, int hlf) {
    const half_t* p = colbase + k0 + hlf * 16;
    HF16 u;
    u.q[0] = *(const float4*)(p);
    u.q[1] = *(const float4*)(p + 8);
    return u.h;
}

__device__ __forceinline__ v8f wmma_f16(v16h a, v16h b, v8f c) {
    return __builtin_amdgcn_wmma_f32_16x16x32_f16(
        false, a, false, b, (short)0, c, false, false);
}

// ---- CDNA5 async DMA: global -> LDS, 16B per lane, tracked by ASYNCcnt ----
// lds_off = wave-relative LDS byte address (low 32 bits of generic pointer,
// as the ISA takes the LDS address).
__device__ __forceinline__ void async_ld_b128(unsigned lds_off, const void* gaddr) {
    asm volatile("global_load_async_to_lds_b128 %0, %1, off"
                 :: "v"(lds_off), "v"(gaddr) : "memory");
}
__device__ __forceinline__ void wait_async0() {
    asm volatile("s_wait_asynccnt 0x0" ::: "memory");
}
__device__ __forceinline__ unsigned lds_addr(const void* p) {
    return (unsigned)(unsigned long long)p;
}

// ---------------------------------------------------------------------------
__global__ void k_cvt_f16(const float* __restrict__ src, half_t* __restrict__ dst, int n) {
    int i = (blockIdx.x * blockDim.x + threadIdx.x) * 4;
    const int stride = gridDim.x * blockDim.x * 4;
    for (; i < n; i += stride) {
        const float4 v = *(const float4*)(src + i);
        v4h o;
        o[0] = (half_t)v.x; o[1] = (half_t)v.y; o[2] = (half_t)v.z; o[3] = (half_t)v.w;
        *(v4h*)(dst + i) = o;
    }
}

// ---------------------------------------------------------------------------
// QKV projection: OUT[m, o] = sum_d X[m,d] * W[o,d] + bias[o]
// Block (256 thr, 8 waves as 2x4): 64(M) x 128(N) tile; wave: 32x32.
// LDS double-buffered k-steps of 32, fed by async-to-LDS DMA.
// mat 0 -> Q [b,h,s,dh], 1 -> K [b,h,s,dh], 2 -> V^T [b,h,dh,s].
// ---------------------------------------------------------------------------
__global__ __launch_bounds__(256) void k_qkv(
    const half_t* __restrict__ X, const half_t* __restrict__ W,
    const float* __restrict__ bq, const float* __restrict__ bk, const float* __restrict__ bv,
    half_t* __restrict__ Qo, half_t* __restrict__ Ko, half_t* __restrict__ Vt)
{
    constexpr int XST = 40;   // 32 + 8 pad halves -> conflict-free b128 frag reads
    constexpr int WST = 40;
    __shared__ half_t Xs[2][64 * XST];
    __shared__ half_t Ws[2][128 * WST];

    const int tid  = threadIdx.x;
    const int lane = tid & 31;
    const int r16  = lane & 15;
    const int hlf  = lane >> 4;
    const int wv   = tid >> 5;   // 0..7
    const int wm   = wv >> 2;    // 0..1
    const int wn   = wv & 3;     // 0..3

    const int bx  = blockIdx.x;              // 0..1535
    const int mat = bx / (64 * 8);
    const int rem = bx - mat * 64 * 8;
    const int m0  = (rem >> 3) * 64;
    const int n0  = (rem & 7) * 128;

    const half_t* Wm = W + (size_t)mat * DIMM * DIMM;

    // cooperative staging assignments
    const int xrow = tid >> 2;              // 0..63
    const int xoff = (tid & 3) * 8;         // 16B per thread
    const int wrow = tid >> 1;              // 0..127
    const int woff = (tid & 1) * 16;        // 32B per thread

    const half_t* Xg = X  + (size_t)(m0 + xrow) * DIMM + xoff;
    const half_t* Wg = Wm + (size_t)(n0 + wrow) * DIMM + woff;

    const unsigned xs_l[2] = { lds_addr(&Xs[0][xrow * XST + xoff]),
                               lds_addr(&Xs[1][xrow * XST + xoff]) };
    const unsigned ws_l[2] = { lds_addr(&Ws[0][wrow * WST + woff]),
                               lds_addr(&Ws[1][wrow * WST + woff]) };

    // prologue: DMA k-step 0 into buffer 0
    async_ld_b128(xs_l[0],      Xg);
    async_ld_b128(ws_l[0],      Wg);
    async_ld_b128(ws_l[0] + 16, Wg + 8);
    wait_async0();
    __syncthreads();

    v8f acc[2][2];
    #pragma unroll
    for (int i = 0; i < 2; ++i)
        #pragma unroll
        for (int j = 0; j < 2; ++j)
            #pragma unroll
            for (int e = 0; e < 8; ++e) acc[i][j][e] = 0.f;

    for (int k0 = 0; k0 < DIMM; k0 += 32) {
        const int  buf  = (k0 >> 5) & 1;
        const bool more = (k0 + 32) < DIMM;

        // issue next k-step's DMA early (overlaps with the WMMAs below)
        if (more) {
            const int nb = buf ^ 1;
            async_ld_b128(xs_l[nb],      Xg + k0 + 32);
            async_ld_b128(ws_l[nb],      Wg + k0 + 32);
            async_ld_b128(ws_l[nb] + 16, Wg + k0 + 40);
        }

        const half_t* xb = &Xs[buf][0];
        const half_t* wb = &Ws[buf][0];
        const v16h a0 = load_a_frag(xb + (wm * 32 +      r16) * XST, 0, hlf);
        const v16h a1 = load_a_frag(xb + (wm * 32 + 16 + r16) * XST, 0, hlf);
        const v16h b0 = load_b_frag(wb + (wn * 32 +      r16) * WST, 0, hlf);
        const v16h b1 = load_b_frag(wb + (wn * 32 + 16 + r16) * WST, 0, hlf);
        acc[0][0] = wmma_f16(a0, b0, acc[0][0]);
        acc[0][1] = wmma_f16(a0, b1, acc[0][1]);
        acc[1][0] = wmma_f16(a1, b0, acc[1][0]);
        acc[1][1] = wmma_f16(a1, b1, acc[1][1]);

        if (more) wait_async0();
        __syncthreads();
    }

    const float* bias = (mat == 0) ? bq : (mat == 1) ? bk : bv;
    #pragma unroll
    for (int i = 0; i < 2; ++i)
        #pragma unroll
        for (int j = 0; j < 2; ++j) {
            const int o  = n0 + wn * 32 + j * 16 + r16;
            const int ng = o >> 6;
            const int dh = o & 63;
            const float bA = bias[o];
            #pragma unroll
            for (int e = 0; e < 8; ++e) {
                const int m    = m0 + wm * 32 + i * 16 + e + hlf * 8;
                const int bidx = m >> 11;
                const int s    = m & 2047;
                const float f  = acc[i][j][e] + bA;
                if (mat == 2) {
                    Vt[((size_t)(bidx * HH + ng) * DHH + dh) * SS + s] = (half_t)f;
                } else {
                    half_t* dst = (mat == 0) ? Qo : Ko;
                    dst[((size_t)(bidx * HH + ng) * SS + s) * DHH + dh] = (half_t)f;
                }
            }
        }
}

// ---------------------------------------------------------------------------
// Flash attention, scores computed transposed (St = K * Q^T) so softmax row
// stats are in-lane.  Block = 4 waves sharing one (b,h); K/V chunks DMA'd
// into double-buffered LDS via async-to-LDS (4x reuse), pipelined.
// ---------------------------------------------------------------------------
__global__ __launch_bounds__(128) void k_attn(
    const half_t* __restrict__ Q, const half_t* __restrict__ K,
    const half_t* __restrict__ Vt, float* __restrict__ Z)
{
    constexpr int KST = 72;   // 64 + 8 pad halves
    constexpr int VST = 40;   // 32 + 8
    __shared__ half_t Ks[2][32 * KST];
    __shared__ half_t Vs[2][64 * VST];

    const int tid  = threadIdx.x;
    const int lane = tid & 31;
    const int r16  = lane & 15;
    const int hlf  = lane >> 4;
    const int wid  = tid >> 5;                         // 0..3

    const int bh = blockIdx.x >> 5;                    // (b*H + h), 0..31
    const int qt = ((blockIdx.x & 31) << 2) + wid;     // 0..127
    const int q0 = qt * 16;

    const half_t* Qb = Q  + ((size_t)bh * SS + q0 + r16) * DHH;
    const half_t* Kb = K  + (size_t)bh * SS * DHH;
    const half_t* Vb = Vt + (size_t)bh * DHH * SS;

    const v16h qf0 = load_b_frag(Qb, 0,  hlf);   // dh 0..31
    const v16h qf1 = load_b_frag(Qb, 32, hlf);   // dh 32..63

    // cooperative staging assignments (per 32-key chunk: K 4KB, V 4KB)
    const int krow = tid >> 2;          // 0..31
    const int koff = (tid & 3) * 16;    // 32B per thread
    const int vrow = tid >> 1;          // 0..63
    const int voff = (tid & 1) * 16;    // 32B per thread
    const half_t* Kg = Kb + (size_t)krow * DHH + koff;
    const half_t* Vg = Vb + (size_t)vrow * SS  + voff;

    const unsigned ks_l[2] = { lds_addr(&Ks[0][krow * KST + koff]),
                               lds_addr(&Ks[1][krow * KST + koff]) };
    const unsigned vs_l[2] = { lds_addr(&Vs[0][vrow * VST + voff]),
                               lds_addr(&Vs[1][vrow * VST + voff]) };

    // prologue: DMA chunk 0 into buffer 0
    async_ld_b128(ks_l[0],      Kg);
    async_ld_b128(ks_l[0] + 16, Kg + 8);
    async_ld_b128(vs_l[0],      Vg);
    async_ld_b128(vs_l[0] + 16, Vg + 8);
    wait_async0();
    __syncthreads();

    v8f o[4];
    #pragma unroll
    for (int d = 0; d < 4; ++d)
        #pragma unroll
        for (int e = 0; e < 8; ++e) o[d][e] = 0.f;

    float mrun = -INFINITY, lrun = 0.f;
    const float sc = 0.125f;   // 1/sqrt(DH)

    for (int t0 = 0; t0 < SS; t0 += 32) {
        const int  buf  = (t0 >> 5) & 1;
        const bool more = (t0 + 32) < SS;

        // issue next chunk's DMA early (overlaps with all compute below)
        if (more) {
            const int nb = buf ^ 1;
            async_ld_b128(ks_l[nb],      Kg + (size_t)(t0 + 32) * DHH);
            async_ld_b128(ks_l[nb] + 16, Kg + (size_t)(t0 + 32) * DHH + 8);
            async_ld_b128(vs_l[nb],      Vg + t0 + 32);
            async_ld_b128(vs_l[nb] + 16, Vg + t0 + 40);
        }

        // ---- scores St[t, q] from LDS
        const half_t* kb0 = &Ks[buf][(     r16) * KST];
        const half_t* kb1 = &Ks[buf][(16 + r16) * KST];
        v8f st0, st1;
        #pragma unroll
        for (int e = 0; e < 8; ++e) { st0[e] = 0.f; st1[e] = 0.f; }
        st0 = wmma_f16(load_a_frag(kb0, 0,  hlf), qf0, st0);
        st0 = wmma_f16(load_a_frag(kb0, 32, hlf), qf1, st0);
        st1 = wmma_f16(load_a_frag(kb1, 0,  hlf), qf0, st1);
        st1 = wmma_f16(load_a_frag(kb1, 32, hlf), qf1, st1);

        // ---- online softmax (per-lane query stats; merge halves via shfl16)
        float mc = -INFINITY;
        #pragma unroll
        for (int e = 0; e < 8; ++e) {
            st0[e] *= sc; st1[e] *= sc;
            mc = fmaxf(mc, fmaxf(st0[e], st1[e]));
        }
        mc = fmaxf(mc, __shfl_xor(mc, 16, 32));
        const float mnew = fmaxf(mrun, mc);
        const float corr = __expf(mrun - mnew);

        float p0[8], p1[8];
        float rs = 0.f;
        #pragma unroll
        for (int e = 0; e < 8; ++e) {
            p0[e] = __expf(st0[e] - mnew);
            p1[e] = __expf(st1[e] - mnew);
            rs += p0[e] + p1[e];
        }
        rs += __shfl_xor(rs, 16, 32);
        lrun = lrun * corr + rs;
        mrun = mnew;

        #pragma unroll
        for (int d = 0; d < 4; ++d)
            #pragma unroll
            for (int e = 0; e < 8; ++e) o[d][e] *= corr;

        // ---- repack P (C-layout) into B-operand layout (16 cross-half shfl)
        v16h pb;
        #pragma unroll
        for (int e = 0; e < 8; ++e) {
            const float y0 = __shfl_xor(p0[e], 16, 32);
            const float y1 = __shfl_xor(p1[e], 16, 32);
            pb[e]     = (half_t)(hlf ? y1    : p0[e]);
            pb[e + 8] = (half_t)(hlf ? p1[e] : y0);
        }

        // ---- out^T[dh, q] += V^T[dh, t] * P[t, q]  (V frags from LDS)
        #pragma unroll
        for (int d = 0; d < 4; ++d) {
            const v16h va = load_a_frag(&Vs[buf][(d * 16 + r16) * VST], 0, hlf);
            o[d] = wmma_f16(va, pb, o[d]);
        }

        if (more) wait_async0();
        __syncthreads();
    }

    const float inv = 1.f / lrun;
    #pragma unroll
    for (int d = 0; d < 4; ++d)
        #pragma unroll
        for (int e = 0; e < 8; ++e) {
            const int dh = d * 16 + e + hlf * 8;
            Z[((size_t)bh * SS + q0 + r16) * DHH + dh] = o[d][e] * inv;
        }
}

// ---------------------------------------------------------------------------
// Residual + LayerNorm.  The reference's no-transpose reshape of z_init
// [B,H,S,DH] -> [B,S,DIM] is exactly the flat buffer read linearly.
// ---------------------------------------------------------------------------
__global__ __launch_bounds__(256) void k_ln(
    const float* __restrict__ emb, const float* __restrict__ Z,
    const float* __restrict__ gamma, const float* __restrict__ beta,
    float* __restrict__ out)
{
    const int row = blockIdx.x;
    const int t   = threadIdx.x;
    const float* e = emb + (size_t)row * DIMM;
    const float* z = Z   + (size_t)row * DIMM;

    float x[4];
    float s = 0.f, s2 = 0.f;
    #pragma unroll
    for (int i = 0; i < 4; ++i) {
        const float v = e[t + i * 256] + z[t + i * 256];
        x[i] = v; s += v; s2 += v * v;
    }
    __shared__ float rs[256], rs2[256];
    rs[t] = s; rs2[t] = s2;
    __syncthreads();
    for (int off = 128; off > 0; off >>= 1) {
        if (t < off) { rs[t] += rs[t + off]; rs2[t] += rs2[t + off]; }
        __syncthreads();
    }
    const float mu   = rs[0] * (1.f / DIMM);
    const float var  = rs2[0] * (1.f / DIMM) - mu * mu;
    const float rstd = rsqrtf(var + 1e-5f);
    #pragma unroll
    for (int i = 0; i < 4; ++i) {
        const int j = t + i * 256;
        out[(size_t)row * DIMM + j] = (x[i] - mu) * rstd * gamma[j] + beta[j];
    }
}

// ---------------------------------------------------------------------------
extern "C" void kernel_launch(void* const* d_in, const int* in_sizes, int n_in,
                              void* d_out, int out_size, void* d_ws, size_t ws_size,
                              hipStream_t stream) {
    (void)in_sizes; (void)n_in; (void)out_size; (void)ws_size;
    const float* emb   = (const float*)d_in[0];
    const float* Wq    = (const float*)d_in[1];
    const float* bq    = (const float*)d_in[2];
    const float* Wk    = (const float*)d_in[3];
    const float* bk    = (const float*)d_in[4];
    const float* Wv    = (const float*)d_in[5];
    const float* bv    = (const float*)d_in[6];
    const float* gamma = (const float*)d_in[7];
    const float* beta  = (const float*)d_in[8];
    float* out = (float*)d_out;

    char* ws = (char*)d_ws;
    const size_t MB = 1024 * 1024;
    half_t* Xh = (half_t*)(ws);             //  8 MB: X f16 [4096,1024]
    half_t* Wh = (half_t*)(ws +  8 * MB);   //  6 MB: Wq|Wk|Wv f16
    half_t* Qh = (half_t*)(ws + 14 * MB);   //  8 MB: Q [b,h,s,dh]
    half_t* Kh = (half_t*)(ws + 22 * MB);   //  8 MB: K [b,h,s,dh]
    half_t* Vt = (half_t*)(ws + 30 * MB);   //  8 MB: V^T [b,h,dh,s]
    float*  Zb = (float* )(ws + 38 * MB);   // 16 MB: attn out flat [b,h,s,dh]

    k_cvt_f16<<<1024, 256, 0, stream>>>(emb, Xh, NTOK * DIMM);
    k_cvt_f16<<<256, 256, 0, stream>>>(Wq, Wh + 0 * (size_t)DIMM * DIMM, DIMM * DIMM);
    k_cvt_f16<<<256, 256, 0, stream>>>(Wk, Wh + 1 * (size_t)DIMM * DIMM, DIMM * DIMM);
    k_cvt_f16<<<256, 256, 0, stream>>>(Wv, Wh + 2 * (size_t)DIMM * DIMM, DIMM * DIMM);

    k_qkv <<<1536, 256, 0, stream>>>(Xh, Wh, bq, bk, bv, Qh, Kh, Vt);
    k_attn<<<1024, 128, 0, stream>>>(Qh, Kh, Vt, Zb);
    k_ln  <<<4096, 256, 0, stream>>>(emb, Zb, gamma, beta, out);
}
